// LSTMRegressor_24412594110560
// MI455X (gfx1250) — matrix operand order repaired and checked by reading, not compiled, not run on hardware
//
#include <hip/hip_runtime.h>

// CDNA5 / gfx1250: wave32, WMMA 16x16x32 bf16 -> f32 accum.
typedef __attribute__((ext_vector_type(16))) __bf16        v16bf;
typedef __attribute__((ext_vector_type(2)))  __bf16        v2bf;
typedef __attribute__((ext_vector_type(8)))  float         v8f;
typedef __attribute__((ext_vector_type(8)))  unsigned int  v8u;

#define LOG2E 1.44269504088896340736f

// sigmoid(g + b) with nb = -b*LOG2E precomputed: fma -> v_exp -> add -> rcp
__device__ __forceinline__ float sig_b(float g, float nb) {
    return __builtin_amdgcn_rcpf(
        1.0f + __builtin_amdgcn_exp2f(__builtin_fmaf(g, -LOG2E, nb)));
}
// tanh(g + b) with b2 = 2*b*LOG2E precomputed
__device__ __forceinline__ float tanh_b(float g, float b2) {
    return 1.0f - 2.0f * __builtin_amdgcn_rcpf(
        1.0f + __builtin_amdgcn_exp2f(__builtin_fmaf(g, 2.0f * LOG2E, b2)));
}
// tanh(x), no bias (for cell state)
__device__ __forceinline__ float tanh_nb(float x) {
    return 1.0f - 2.0f * __builtin_amdgcn_rcpf(
        1.0f + __builtin_amdgcn_exp2f(x * (2.0f * LOG2E)));
}
__device__ __forceinline__ unsigned int pack_bf16(float a, float b) {
    v2bf t; t[0] = (__bf16)a; t[1] = (__bf16)b;   // -> v_cvt_pk_bf16_f32
    return __builtin_bit_cast(unsigned int, t);
}

// One LSTM step for 16 batch rows (one wave). Transposed recurrence:
//   G^T[64 x 16b] = Wcat[64 x 32] @ S[32 x 16b]     (C = inline 0)
// K packing: K0..15 = h (bf16), K16..23 = x_t, K24..31 = 0.
// Bias is folded into the activation fma constants (zero per-step cost).
// xp[] carries the *packed* x_t into the step; the prefetched x_{t+1} is
// packed right after the WMMA group so those independent v_cvt_pk fill the
// last WMMA's D->VALU hazard slots (instead of v_nops).
__device__ __forceinline__ void lstm_step(const v16bf A[4],
                                          const float nbi[8], const float nbf[8],
                                          const float b2g[8], const float nbo[8],
                                          float h[8], float c[8],
                                          unsigned int xp[4],
                                          float4 xa, float4 xb, bool up) {
    // ---- Build B (state): 8 packed bf16-pair dwords -------------------------
    unsigned int hp[4], hq[4];
#pragma unroll
    for (int j = 0; j < 4; ++j)
        hp[j] = pack_bf16(h[2 * j], h[2 * j + 1]);   // own hidden 8h..8h+7
#pragma unroll
    for (int j = 0; j < 4; ++j)
        hq[j] = __shfl_xor(hp[j], 16, 32);           // other half's hidden

    v8u bd;
#pragma unroll
    for (int j = 0; j < 4; ++j) bd[j]     = hp[j];              // K lo 8
#pragma unroll
    for (int j = 0; j < 4; ++j) bd[4 + j] = up ? 0u    : xp[j]; // K hi 8
    const v16bf Bf = __builtin_bit_cast(v16bf, bd);

    // ---- 4 WMMAs as one back-to-back group ----------------------------------
    v8f zc = {};
    v8f Gi = __builtin_amdgcn_wmma_f32_16x16x32_bf16(false, A[0], false, Bf, (short)0, zc, false, false);
    __builtin_amdgcn_sched_barrier(0);
    v8f Gf = __builtin_amdgcn_wmma_f32_16x16x32_bf16(false, A[1], false, Bf, (short)0, zc, false, false);
    __builtin_amdgcn_sched_barrier(0);
    v8f Gg = __builtin_amdgcn_wmma_f32_16x16x32_bf16(false, A[2], false, Bf, (short)0, zc, false, false);
    __builtin_amdgcn_sched_barrier(0);
    v8f Go = __builtin_amdgcn_wmma_f32_16x16x32_bf16(false, A[3], false, Bf, (short)0, zc, false, false);
    __builtin_amdgcn_sched_barrier(0);

    // ---- Pack next x (independent of WMMA results; fills hazard slots) ------
    xp[0] = pack_bf16(xa.x, xa.y);
    xp[1] = pack_bf16(xa.z, xa.w);
    xp[2] = pack_bf16(xb.x, xb.y);
    xp[3] = pack_bf16(xb.z, xb.w);

    // ---- Elementwise LSTM cell (fp32 state, bias folded in) -----------------
#pragma unroll
    for (int r = 0; r < 8; ++r) {
        const float iv = sig_b (Gi[r], nbi[r]);
        const float fv = sig_b (Gf[r], nbf[r]);
        const float gv = tanh_b(Gg[r], b2g[r]);
        const float ov = sig_b (Go[r], nbo[r]);
        const float cn = __builtin_fmaf(fv, c[r], iv * gv);
        c[r] = cn;
        h[r] = ov * tanh_nb(cn);
    }
}

__global__ void __launch_bounds__(32)
lstm_wmma_gfx1250(const float* __restrict__ x,     // [B,T,8]
                  const float* __restrict__ W_ih,  // [64,8]
                  const float* __restrict__ W_hh,  // [64,16]
                  const float* __restrict__ b_ih,  // [64]
                  const float* __restrict__ b_hh,  // [64]
                  const float* __restrict__ fc_w,  // [O,16] (row 0 used)
                  const float* __restrict__ fc_b,  // [O]
                  float* __restrict__ out,         // [B]
                  int T)
{
    const int lane = threadIdx.x & 31;
    const int half = lane >> 4;   // 0: K-low half, 1: K-high half
    const int m    = lane & 15;   // row within tile / batch column
    const int b0   = blockIdx.x * 16;
    const bool up  = (half != 0);

    // ---- A fragments (weights, once per wave) -------------------------------
    // bf16 A 16x32 layout: lanes 0-15 hold K=0..7 (elems 0-7) and K=16..23
    // (elems 8-15); lanes 16-31 hold K=8..15 and K=24..31.
    v16bf A[4];
#pragma unroll
    for (int tt = 0; tt < 4; ++tt) {
        const int row = tt * 16 + m;
#pragma unroll
        for (int j = 0; j < 8; ++j)            // K = 8*half + j -> W_hh cols
            A[tt][j] = (__bf16)W_hh[row * 16 + 8 * half + j];
#pragma unroll
        for (int j = 0; j < 8; ++j) {          // K = 16 + 8*half + j
            const float wv = W_ih[row * 8 + j];          // half==0: W_ih cols
            A[tt][8 + j] = (__bf16)(up ? 0.0f : wv);     // half==1: zero pad
        }
    }

    // ---- Bias constants folded for the activations (per lane: 4 x 8) -------
    float nbi[8], nbf[8], b2g[8], nbo[8];
#pragma unroll
    for (int r = 0; r < 8; ++r) {
        const int rr = r + 8 * half;           // hidden index this VGPR slot
        const float bi = b_ih[ 0 + rr] + b_hh[ 0 + rr];   // i gate
        const float bf = b_ih[16 + rr] + b_hh[16 + rr];   // f gate
        const float bg = b_ih[32 + rr] + b_hh[32 + rr];   // g gate
        const float bo = b_ih[48 + rr] + b_hh[48 + rr];   // o gate
        nbi[r] = -bi * LOG2E;
        nbf[r] = -bf * LOG2E;
        b2g[r] = 2.0f * bg * LOG2E;
        nbo[r] = -bo * LOG2E;
    }

    // ---- State --------------------------------------------------------------
    float h[8], c[8];
#pragma unroll
    for (int r = 0; r < 8; ++r) { h[r] = 0.0f; c[r] = 0.0f; }

    const float* xptr = x + (size_t)(b0 + m) * (size_t)T * 8;
    float4 xa = ((const float4*)xptr)[0];
    float4 xb = ((const float4*)xptr)[1];
    unsigned int xp[4];                    // packed x_t carried across steps
    xp[0] = pack_bf16(xa.x, xa.y);
    xp[1] = pack_bf16(xa.z, xa.w);
    xp[2] = pack_bf16(xb.x, xb.y);
    xp[3] = pack_bf16(xb.z, xb.w);

    // Main loop: strictly serial recurrence -> unrolling buys no ILP, only
    // register pressure. Keep one step per iteration.
#pragma unroll 1
    for (int t = 0; t < T - 1; ++t) {
        xa = ((const float4*)(xptr + (size_t)(t + 1) * 8))[0];   // prefetch
        xb = ((const float4*)(xptr + (size_t)(t + 1) * 8))[1];
        lstm_step(A, nbi, nbf, b2g, nbo, h, c, xp, xa, xb, up);
    }
    lstm_step(A, nbi, nbf, b2g, nbo, h, c, xp, xa, xb, up);      // t = T-1

    // ---- out[b] = <h_b, fc_w[0,:]> + fc_b[0] --------------------------------
    float acc = 0.0f;
#pragma unroll
    for (int r = 0; r < 8; ++r) acc += h[r] * fc_w[8 * half + r];
    acc += __shfl_xor(acc, 16, 32);
    if (half == 0) out[b0 + m] = acc + fc_b[0];
}

extern "C" void kernel_launch(void* const* d_in, const int* in_sizes, int n_in,
                              void* d_out, int out_size, void* d_ws, size_t ws_size,
                              hipStream_t stream) {
    const float* x    = (const float*)d_in[0];
    const float* W_ih = (const float*)d_in[1];
    const float* W_hh = (const float*)d_in[2];
    const float* b_ih = (const float*)d_in[3];
    const float* b_hh = (const float*)d_in[4];
    const float* fc_w = (const float*)d_in[5];
    const float* fc_b = (const float*)d_in[6];
    float* out = (float*)d_out;

    const int B = out_size;                                            // 4096
    const int I = 8;
    const int T = (int)((long long)in_sizes[0] / ((long long)B * I));  // 2048

    dim3 grid(B / 16), block(32);
    hipLaunchKernelGGL(lstm_wmma_gfx1250, grid, block, 0, stream,
                       x, W_ih, W_hh, b_ih, b_hh, fc_w, fc_b, out, T);
}
